// NHLayer_60215441490491
// MI455X (gfx1250) — matrix operand order repaired and checked by reading, not compiled, not run on hardware
//
#include <hip/hip_runtime.h>

#define HN 64
#define CIN 8
#define HH 256
#define WW 256
#define EE 4
#define DD 4

#define TS 16
#define ZW 20            // 16 + 2*2 halo
#define ZA (ZW*ZW)       // 400
#define H1W 18           // 16 + 2*1 halo
#define H1A (H1W*H1W)    // 324
#define GP  336          // 21 groups * 16 px: padded pixel count (tail-safe)

#define KP_E  64         // padded K for 4-ch 3x3 convs (K=36)
#define KP_N1 128        // padded K for 12-ch 3x3 conv (K=108)

typedef __attribute__((ext_vector_type(16))) _Float16 v16h;
typedef __attribute__((ext_vector_type(8)))  _Float16 v8h;
typedef __attribute__((ext_vector_type(4)))  _Float16 v4h;
typedef __attribute__((ext_vector_type(8)))  float    v8f;

__device__ __forceinline__ float silu_f(float v) {
  // v * sigmoid(v); rcp instead of IEEE divide (avoids div_scale/div_fixup chain)
  return v * __builtin_amdgcn_rcpf(1.f + __expf(-v));
}

// ---- A operand: pixel p's K-slice [32*s, 32*s+32) from K-contiguous im2col.
// A k order chosen here (contiguous, same as B): lane holds M=lane&15; halves 0..15 = K{16hi..16hi+15}
// (matches B packing)  ->  two aligned 16-byte LDS loads.
__device__ __forceinline__ v16h load_A(const _Float16* __restrict__ icol,
                                       int base_off, int hlane) {
  const v8h lo = *(const v8h*)(icol + base_off + 16*hlane);
  const v8h up = *(const v8h*)(icol + base_off + 16*hlane + 8);
  return __builtin_shufflevector(lo, up, 0,1,2,3,4,5,6,7,8,9,10,11,12,13,14,15);
}

// ---- B operand: (K x 16) weights, out-channels in N (n<4 valid), packed once.
// B k order chosen here (contiguous): lanes 0-15 hold N=lane, halves 0..15 = K 0..15;
// lanes 16-31 hold N=lane-16, halves 0..15 = K 16..31.
// K ordering matches im2col: k = (dy*3+dx)*C + ci.
template<int C>
__device__ __forceinline__ v16h pack_Bw(const float* __restrict__ w, int lane, int step) {
  const int n  = lane & 15;
  const int hi = lane >> 4;
  v16h b;
#pragma unroll
  for (int h = 0; h < 16; ++h) {
    const int k = 32*step + 16*hi + h;
    float v = 0.f;
    if (n < 4 && k < C*9) {
      const int t = k / C, ci = k % C;
      v = w[((n*C + ci)*3 + (t/3))*3 + (t%3)];
    }
    b[h] = (_Float16)v;
  }
  return b;
}

// ================= Kernel 1: down + edge CNN + cross-cell sum =================
__global__ __launch_bounds__(256) void edge_kernel(
    const float* __restrict__ x,
    const float* __restrict__ w_down, const float* __restrict__ b_down,
    const float* __restrict__ w_e1,   const float* __restrict__ b_e1,
    const float* __restrict__ w_e2,   const float* __restrict__ b_e2,
    float* __restrict__ Aout) {
  __shared__ __align__(16) _Float16 zt[ZA*EE];     // [20x20 pixel][4ch]
  __shared__ __align__(16) _Float16 h1t[GP*EE];    // [18x18 pixel][4ch], tail-padded
  __shared__ __align__(16) _Float16 icol[GP*KP_E]; // im2col, reused e1/e2, tail-padded

  const int tid   = threadIdx.x;
  const int lane  = tid & 31, wv = tid >> 5;
  const int nlane = lane & 15, hlane = lane >> 4;
  const int tx0 = blockIdx.x * TS, ty0 = blockIdx.y * TS;

  float wd[EE*CIN], bd[EE];
#pragma unroll
  for (int i = 0; i < EE*CIN; ++i) wd[i] = w_down[i];
#pragma unroll
  for (int c = 0; c < EE; ++c) bd[c] = b_down[c];

  v16h Be1[2], Be2[2];
#pragma unroll
  for (int s = 0; s < 2; ++s) { Be1[s] = pack_Bw<EE>(w_e1, lane, s); Be2[s] = pack_Bw<EE>(w_e2, lane, s); }

  const float bias1 = (nlane < EE) ? b_e1[nlane] : 0.f;
  const float bias2 = (nlane < EE) ? b_e2[nlane] : 0.f;

  // register-resident cross-cell accumulator: wave wv owns groups wv and wv+8
  v8f asum[2];
#pragma unroll
  for (int j = 0; j < 2; ++j)
#pragma unroll
    for (int r = 0; r < 8; ++r) asum[j][r] = 0.f;

  // zero icol once: covers K-pad [36,64) and pixel-pad [324,336); builds write k<36, p<324
  {
    const v4h z4 = {};
    for (int i = tid; i < GP*KP_E/4; i += 256) *(v4h*)(icol + i*4) = z4;
  }
  __syncthreads();

  for (int n = 0; n < HN; ++n) {
    // ---- stage z = down(x), f16, 20x20 halo tile, pixel-interleaved
    for (int i = tid; i < ZA; i += 256) {
      const int r = i / ZW, c = i % ZW;
      const int gy = ty0 - 2 + r, gx = tx0 - 2 + c;
      const bool inb = (gy >= 0) & (gy < HH) & (gx >= 0) & (gx < WW);
      float xi[CIN];
#pragma unroll
      for (int ci = 0; ci < CIN; ++ci)
        xi[ci] = inb ? x[(((size_t)n*CIN + ci)*HH + gy)*WW + gx] : 0.f;
      v4h q = {};
#pragma unroll
      for (int co = 0; co < EE; ++co) {
        float s = bd[co];
#pragma unroll
        for (int ci = 0; ci < CIN; ++ci) s += wd[co*CIN + ci] * xi[ci];
        q[co] = (_Float16)s;
      }
      *(v4h*)(zt + i*EE) = q;
    }
    __syncthreads();

    // ---- im2col for e1: 324 px, 9 taps x 4ch (b64 copies)
    for (int p = tid; p < H1A; p += 256) {
      const int row = p / H1W, col = p % H1W;
      const _Float16* sp = zt + (row*ZW + col)*EE;
      _Float16* dp = icol + p*KP_E;
#pragma unroll
      for (int t = 0; t < 9; ++t)
        *(v4h*)(dp + t*EE) = *(const v4h*)(sp + ((t/3)*ZW + (t%3))*EE);
    }
    __syncthreads();

    // ---- h1 = silu(e1 * z) on 18x18 (21 groups of 16 px; pads read zero rows; their outputs are unused)
    for (int g = wv; g < GP/16; g += 8) {
      const int base = g*16;
      const int p = base + nlane;
      v8f acc;
#pragma unroll
      for (int r = 0; r < 8; ++r) acc[r] = bias1;
#pragma unroll
      for (int s = 0; s < 2; ++s) {
        v16h a = load_A(icol, p*KP_E + 32*s, hlane);
        acc = __builtin_amdgcn_wmma_f32_16x16x32_f16(false, a, false, Be1[s], (short)0, acc, false, false);
      }
      if (nlane < EE) {
#pragma unroll
        for (int r = 0; r < 8; ++r) {
          const int pp = base + r + 8*hlane;       // < GP, always in-bounds
          h1t[pp*EE + nlane] = (_Float16)silu_f(acc[r]);
        }
      }
    }
    __syncthreads();

    // ---- im2col for e2: 256 px
    for (int p = tid; p < TS*TS; p += 256) {
      const int row = p / TS, col = p % TS;
      const _Float16* sp = h1t + (row*H1W + col)*EE;
      _Float16* dp = icol + p*KP_E;
#pragma unroll
      for (int t = 0; t < 9; ++t)
        *(v4h*)(dp + t*EE) = *(const v4h*)(sp + ((t/3)*H1W + (t%3))*EE);
    }
    __syncthreads();

    // ---- h2 = silu(e2 * h1), accumulate in registers (deterministic, lane-owned)
#pragma unroll
    for (int j = 0; j < 2; ++j) {
      const int base = (wv + 8*j)*16;
      const int p = base + nlane;
      v8f acc;
#pragma unroll
      for (int r = 0; r < 8; ++r) acc[r] = bias2;
#pragma unroll
      for (int s = 0; s < 2; ++s) {
        v16h a = load_A(icol, p*KP_E + 32*s, hlane);
        acc = __builtin_amdgcn_wmma_f32_16x16x32_f16(false, a, false, Be2[s], (short)0, acc, false, false);
      }
#pragma unroll
      for (int r = 0; r < 8; ++r) asum[j][r] += silu_f(acc[r]);
    }
    __syncthreads();
  }

  // ---- write cross-cell aggregate straight from registers
  if (nlane < EE) {
#pragma unroll
    for (int j = 0; j < 2; ++j) {
      const int base = (wv + 8*j)*16;
#pragma unroll
      for (int r = 0; r < 8; ++r) {
        const int pp = base + r + 8*hlane;
        Aout[((size_t)nlane*HH + ty0 + pp/TS)*WW + tx0 + pp%TS] = asum[j][r];
      }
    }
  }
}

// ========== Kernel 2: up(A) + concat + node CNN + residual ==========
__global__ __launch_bounds__(256) void node_kernel(
    const float* __restrict__ x, const float* __restrict__ Ain,
    const float* __restrict__ w_up, const float* __restrict__ b_up,
    const float* __restrict__ w_n1, const float* __restrict__ b_n1,
    const float* __restrict__ w_n2, const float* __restrict__ b_n2,
    const float* __restrict__ w_r,  const float* __restrict__ b_r,
    float* __restrict__ out) {
  __shared__ __align__(16) _Float16 xat[ZA*(CIN+EE)]; // [20x20 pixel][12ch]
  __shared__ __align__(16) _Float16 tt[GP*DD];        // [18x18 pixel][4ch], tail-padded
  __shared__ __align__(16) _Float16 icol[GP*KP_N1];   // reused n1 (K128) / n2 (K64)

  const int tid   = threadIdx.x;
  const int lane  = tid & 31, wv = tid >> 5;
  const int nlane = lane & 15, hlane = lane >> 4;
  const int tx0 = blockIdx.x * TS, ty0 = blockIdx.y * TS;
  const int n = blockIdx.z;

  float wu[EE*EE], bu[EE];
#pragma unroll
  for (int i = 0; i < EE*EE; ++i) wu[i] = w_up[i];
#pragma unroll
  for (int c = 0; c < EE; ++c) bu[c] = b_up[c];

  v16h Bn1[4], Bn2[2];
#pragma unroll
  for (int s = 0; s < 4; ++s) Bn1[s] = pack_Bw<CIN+EE>(w_n1, lane, s);
#pragma unroll
  for (int s = 0; s < 2; ++s) Bn2[s] = pack_Bw<DD>(w_n2, lane, s);

  const float bias1 = (nlane < DD) ? b_n1[nlane] : 0.f;
  const float bias2 = (nlane < DD) ? b_n2[nlane] : 0.f;

  // zero icol once: covers n1 K-pad [108,128) and pixel-pad [324,336)
  {
    const v4h z4 = {};
    for (int i = tid; i < GP*KP_N1/4; i += 256) *(v4h*)(icol + i*4) = z4;
  }

  // ---- stage concat[x, up(A)] f16, pixel-interleaved; SAME-pad of concat is zero
  for (int i = tid; i < ZA; i += 256) {
    const int r = i / ZW, c = i % ZW;
    const int gy = ty0 - 2 + r, gx = tx0 - 2 + c;
    const bool inb = (gy >= 0) & (gy < HH) & (gx >= 0) & (gx < WW);
    v4h q0 = {}, q1 = {}, q2 = {};
#pragma unroll
    for (int ci = 0; ci < 4; ++ci)
      q0[ci] = (_Float16)(inb ? x[(((size_t)n*CIN + ci)*HH + gy)*WW + gx] : 0.f);
#pragma unroll
    for (int ci = 0; ci < 4; ++ci)
      q1[ci] = (_Float16)(inb ? x[(((size_t)n*CIN + 4 + ci)*HH + gy)*WW + gx] : 0.f);
    float av[EE];
#pragma unroll
    for (int ci = 0; ci < EE; ++ci)
      av[ci] = inb ? Ain[((size_t)ci*HH + gy)*WW + gx] : 0.f;
#pragma unroll
    for (int co = 0; co < EE; ++co) {
      float s = bu[co];
#pragma unroll
      for (int ci = 0; ci < EE; ++ci) s += wu[co*EE + ci] * av[ci];
      q2[co] = (_Float16)(inb ? s : 0.f);
    }
    _Float16* dp = xat + i*(CIN+EE);
    *(v4h*)(dp)     = q0;
    *(v4h*)(dp + 4) = q1;
    *(v4h*)(dp + 8) = q2;
  }
  __syncthreads();

  // ---- im2col for n1: 324 px, 9 taps x 12ch (3x b64 copies per tap)
  for (int p = tid; p < H1A; p += 256) {
    const int row = p / H1W, col = p % H1W;
    const _Float16* sp = xat + (row*ZW + col)*(CIN+EE);
    _Float16* dp = icol + p*KP_N1;
#pragma unroll
    for (int t = 0; t < 9; ++t) {
      const _Float16* s2 = sp + ((t/3)*ZW + (t%3))*(CIN+EE);
      _Float16* d2 = dp + t*(CIN+EE);
      *(v4h*)(d2)     = *(const v4h*)(s2);
      *(v4h*)(d2 + 4) = *(const v4h*)(s2 + 4);
      *(v4h*)(d2 + 8) = *(const v4h*)(s2 + 8);
    }
  }
  __syncthreads();

  // ---- t = silu(n1 * concat): K=108 -> 4 steps, 18x18 region (21 groups)
  for (int g = wv; g < GP/16; g += 8) {
    const int base = g*16;
    const int p = base + nlane;
    v8f acc;
#pragma unroll
    for (int r = 0; r < 8; ++r) acc[r] = bias1;
#pragma unroll
    for (int s = 0; s < 4; ++s) {
      v16h a = load_A(icol, p*KP_N1 + 32*s, hlane);
      acc = __builtin_amdgcn_wmma_f32_16x16x32_f16(false, a, false, Bn1[s], (short)0, acc, false, false);
    }
    if (nlane < DD) {
#pragma unroll
      for (int r = 0; r < 8; ++r) {
        const int pp = base + r + 8*hlane;         // < GP, always in-bounds
        tt[pp*DD + nlane] = (_Float16)silu_f(acc[r]);
      }
    }
  }
  __syncthreads();

  // ---- rebuild icol for n2 (Kpad=64, 256 px): pad [36,64) + 9 tap copies
  {
    const v4h z4 = {};
    for (int i = tid; i < TS*TS*7; i += 256) {
      const int p = i / 7, j = i % 7;
      *(v4h*)(icol + p*KP_E + 36 + j*4) = z4;
    }
  }
  for (int p = tid; p < TS*TS; p += 256) {
    const int row = p / TS, col = p % TS;
    const _Float16* sp = tt + (row*H1W + col)*DD;
    _Float16* dp = icol + p*KP_E;
#pragma unroll
    for (int t = 0; t < 9; ++t)
      *(v4h*)(dp + t*DD) = *(const v4h*)(sp + ((t/3)*H1W + (t%3))*DD);
  }
  __syncthreads();

  // ---- y = n2 * t + (w_r . x + b_r), 16x16 region (16 groups, no tail)
#pragma unroll
  for (int j = 0; j < 2; ++j) {
    const int base = (wv + 8*j)*16;
    const int p = base + nlane;
    v8f acc;
#pragma unroll
    for (int r = 0; r < 8; ++r) acc[r] = bias2;
#pragma unroll
    for (int s = 0; s < 2; ++s) {
      v16h a = load_A(icol, p*KP_E + 32*s, hlane);
      acc = __builtin_amdgcn_wmma_f32_16x16x32_f16(false, a, false, Bn2[s], (short)0, acc, false, false);
    }
    if (nlane < DD) {
#pragma unroll
      for (int r = 0; r < 8; ++r) {
        const int pp = base + r + 8*hlane;
        const int rr = pp / TS, cc = pp % TS;
        float s = b_r[nlane];
#pragma unroll
        for (int ci = 0; ci < CIN; ++ci)
          s += w_r[nlane*CIN + ci] * (float)xat[((rr + 2)*ZW + (cc + 2))*(CIN+EE) + ci];
        out[(((size_t)n*DD + nlane)*HH + ty0 + rr)*WW + tx0 + cc] = acc[r] + s;
      }
    }
  }
}

extern "C" void kernel_launch(void* const* d_in, const int* in_sizes, int n_in,
                              void* d_out, int out_size, void* d_ws, size_t ws_size,
                              hipStream_t stream) {
  (void)in_sizes; (void)n_in; (void)out_size; (void)ws_size;
  const float* x      = (const float*)d_in[0];
  const float* w_down = (const float*)d_in[1];
  const float* b_down = (const float*)d_in[2];
  const float* w_e1   = (const float*)d_in[3];
  const float* b_e1   = (const float*)d_in[4];
  const float* w_e2   = (const float*)d_in[5];
  const float* b_e2   = (const float*)d_in[6];
  const float* w_up   = (const float*)d_in[7];
  const float* b_up   = (const float*)d_in[8];
  const float* w_n1   = (const float*)d_in[9];
  const float* b_n1   = (const float*)d_in[10];
  const float* w_n2   = (const float*)d_in[11];
  const float* b_n2   = (const float*)d_in[12];
  const float* w_r    = (const float*)d_in[13];
  const float* b_r    = (const float*)d_in[14];
  float* A   = (float*)d_ws;            // (E, H, W) cross-cell aggregate, 1 MB
  float* out = (float*)d_out;           // (N, D, H, W)

  dim3 blk(256);
  edge_kernel<<<dim3(WW/TS, HH/TS), blk, 0, stream>>>(
      x, w_down, b_down, w_e1, b_e1, w_e2, b_e2, A);
  node_kernel<<<dim3(WW/TS, HH/TS, HN), blk, 0, stream>>>(
      x, A, w_up, b_up, w_n1, b_n1, w_n2, b_n2, w_r, b_r, out);
}
